// ProbAttention_20925080666787
// MI455X (gfx1250) — matrix operand order repaired and checked by reading, not compiled, not run on hardware
//
#include <hip/hip_runtime.h>
#include <hip/hip_bf16.h>

// ---------------------------------------------------------------------------
// ProbSparse attention (Informer) for MI455X / gfx1250, wave32.
// B=16, L=4096, D=512, H=1, SAMPLE_K = N_TOP = 45.
// Big GEMMs (x@W.T) use v_wmma_f32_16x16x32_bf16 with TDM
// (tensor_load_to_lds) double-buffered LDS tile staging.
// ---------------------------------------------------------------------------

typedef __attribute__((ext_vector_type(16))) __bf16    v16bf;
typedef __attribute__((ext_vector_type(8)))  __bf16    v8bf;
typedef __attribute__((ext_vector_type(8)))  float     v8f;
typedef __attribute__((ext_vector_type(4)))  unsigned  v4u;
typedef __attribute__((ext_vector_type(8)))  int       v8i;
typedef __attribute__((ext_vector_type(4)))  int       v4i;

#define NB 16
#define NL 4096
#define ND 512
#define SK 45
#define MROWS (NB * NL)   // 65536
#define SCALE 0.04419417382415922f  // 1/sqrt(512)

#if defined(__has_builtin)
#if __has_builtin(__builtin_amdgcn_tensor_load_to_lds)
#define PA_HAS_TDM 1
#endif
#endif
#ifndef PA_HAS_TDM
#define PA_HAS_TDM 0
#endif

// ------------------------------ f32 -> bf16 --------------------------------
__global__ __launch_bounds__(256) void pa_cvt_bf16(const float* __restrict__ s,
                                                   __bf16* __restrict__ d,
                                                   size_t n) {
  size_t i = (size_t)blockIdx.x * 256 + threadIdx.x;
  if (i < n) d[i] = (__bf16)s[i];
}

// ------------------------------ WMMA GEMM ----------------------------------
// C[M x 512] = A[M x 512](bf16) * W[512 x 512](bf16, row = out-feature)^T + bias
// Block tile 128x128, 8 waves (2 n-groups x 4 m-groups), K-step 32.

__device__ inline v16bf pa_frag_a(const __bf16* t, int lane) {
  int r  = lane & 15;
  int kg = (lane >> 4) * 8;
  const __bf16* p = t + r * 32 + kg;
  v16bf v;
#pragma unroll
  for (int i = 0; i < 8; ++i) v[i] = p[i];
#pragma unroll
  for (int i = 0; i < 8; ++i) v[i + 8] = p[16 + i];
  return v;
}

__device__ inline v16bf pa_frag_b(const __bf16* t, int lane) {
  int c  = lane & 15;
  int kg = (lane >> 4) * 8;
  const __bf16* p = t + c * 32 + kg;
  v16bf v;
#pragma unroll
  for (int i = 0; i < 8; ++i) v[i] = p[i];
#pragma unroll
  for (int i = 0; i < 8; ++i) v[i + 8] = p[16 + i];
  return v;
}

#if PA_HAS_TDM
// Issue a TDM load of a 128-row x 32-elem bf16 tile (row stride 512 elems)
// from global `gsrc` into LDS byte offset `lds_off`. D# per ISA 08 §8:
//  group0: count=1, lds_addr, global_addr[56:0], type=2
//  group1: data_size=1(2B), tensor_dim0=512, tensor_dim1=4096,
//          tile_dim0=32, tile_dim1=128, tensor_dim0_stride=512
__device__ inline void pa_tdm_load_tile(const __bf16* gsrc, unsigned lds_off) {
  unsigned long long ga = (unsigned long long)(size_t)gsrc;
  v4u g0;
  g0[0] = 1u;                                   // count=1 valid descriptor
  g0[1] = lds_off;                              // LDS byte address
  g0[2] = (unsigned)ga;                         // global_addr[31:0]
  g0[3] = (unsigned)((ga >> 32) & 0x01FFFFFFu)  // global_addr[56:32]
          | (2u << 30);                         // type=2 ("image")
  v8i g1;
  g1[0] = (int)(1u << 16);      // data_size=1 -> 2 bytes/elem
  g1[1] = (int)(512u << 16);    // tensor_dim0[15:0]=512 (bits 63:48)
  g1[2] = (int)(4096u << 16);   // tensor_dim0[31:16]=0 | tensor_dim1[15:0]=4096
  g1[3] = (int)(32u << 16);     // tensor_dim1[31:16]=0 | tile_dim0=32
  g1[4] = 128;                  // tile_dim1=128 | tile_dim2=0
  g1[5] = 512;                  // tensor_dim0_stride[31:0]=512
  g1[6] = 0;                    // stride[47:32]=0 | tensor_dim1_stride[15:0]=0
  g1[7] = 0;
  v4i z4 = {0, 0, 0, 0};
#if defined(__clang_major__) && (__clang_major__ >= 23)
  v8i z8 = {0, 0, 0, 0, 0, 0, 0, 0};
  __builtin_amdgcn_tensor_load_to_lds(g0, g1, z4, z4, z8, 0);
#else
  __builtin_amdgcn_tensor_load_to_lds(g0, g1, z4, z4, 0);
#endif
}
#endif

__global__ __launch_bounds__(256) void pa_gemm_bf16(const __bf16* __restrict__ A,
                                                    const __bf16* __restrict__ W,
                                                    const float* __restrict__ bias,
                                                    float* __restrict__ C) {
  __shared__ __align__(16) __bf16 sA[2][128 * 32];
  __shared__ __align__(16) __bf16 sB[2][128 * 32];

  const int tid  = threadIdx.x;
  const int lane = tid & 31;
  const int wave = tid >> 5;
  const int m0   = blockIdx.y * 128;
  const int n0   = blockIdx.x * 128;
  const int wm   = (wave & 3) * 32;   // wave's m offset (two 16-row subtiles)
  const int wn   = (wave >> 2) * 64;  // wave's n offset (four 16-col subtiles)

  v8f acc[2][4];
#pragma unroll
  for (int i = 0; i < 2; ++i)
#pragma unroll
    for (int j = 0; j < 4; ++j)
      acc[i][j] = v8f{0.f, 0.f, 0.f, 0.f, 0.f, 0.f, 0.f, 0.f};

#if PA_HAS_TDM
  // -------- TDM double-buffered pipeline: DMA next tile while computing ----
  const unsigned offA[2] = {(unsigned)(size_t)&sA[0][0], (unsigned)(size_t)&sA[1][0]};
  const unsigned offB[2] = {(unsigned)(size_t)&sB[0][0], (unsigned)(size_t)&sB[1][0]};
  int cur = 0;

  if (tid < 32) {  // wave 0 drives the tensor data mover (EXEC ignored by TDM)
    pa_tdm_load_tile(A + (size_t)m0 * ND, offA[0]);
    pa_tdm_load_tile(W + (size_t)n0 * ND, offB[0]);
  }

  for (int kk = 0; kk < ND; kk += 32) {
    if (tid < 32) {
      if (kk + 32 < ND) {
        // prefetch next K-slice into the other buffer (its readers synced
        // at the end of the previous iteration)
        pa_tdm_load_tile(A + (size_t)m0 * ND + kk + 32, offA[cur ^ 1]);
        pa_tdm_load_tile(W + (size_t)n0 * ND + kk + 32, offB[cur ^ 1]);
        __builtin_amdgcn_s_wait_tensorcnt(2);  // current buffer's 2 loads done
      } else {
        __builtin_amdgcn_s_wait_tensorcnt(0);
      }
    }
    __syncthreads();  // publish current LDS tile to all 8 waves

    v16bf af[2];
    af[0] = pa_frag_a(&sA[cur][0] + (wm + 0)  * 32, lane);
    af[1] = pa_frag_a(&sA[cur][0] + (wm + 16) * 32, lane);
    v16bf bf[4];
#pragma unroll
    for (int j = 0; j < 4; ++j)
      bf[j] = pa_frag_b(&sB[cur][0] + (wn + j * 16) * 32, lane);

#pragma unroll
    for (int i = 0; i < 2; ++i)
#pragma unroll
      for (int j = 0; j < 4; ++j)
        acc[i][j] = __builtin_amdgcn_wmma_f32_16x16x32_bf16(
            false, af[i], false, bf[j], (short)0, acc[i][j], false, false);

    __syncthreads();  // all reads of current buffer done before TDM reuses it
    cur ^= 1;
  }
#else
  // ---------------- fallback: manual LDS staging (single buffer) -----------
  const int lr = tid >> 1;
  const int lh = (tid & 1) * 16;
  for (int kk = 0; kk < ND; kk += 32) {
    const __bf16* gA = A + (size_t)(m0 + lr) * ND + kk + lh;
    *(v8bf*)(&sA[0][0] + lr * 32 + lh)     = *(const v8bf*)(gA);
    *(v8bf*)(&sA[0][0] + lr * 32 + lh + 8) = *(const v8bf*)(gA + 8);
    const __bf16* gW = W + (size_t)(n0 + lr) * ND + kk + lh;
    *(v8bf*)(&sB[0][0] + lr * 32 + lh)     = *(const v8bf*)(gW);
    *(v8bf*)(&sB[0][0] + lr * 32 + lh + 8) = *(const v8bf*)(gW + 8);
    __syncthreads();

    v16bf af[2];
    af[0] = pa_frag_a(&sA[0][0] + (wm + 0)  * 32, lane);
    af[1] = pa_frag_a(&sA[0][0] + (wm + 16) * 32, lane);
    v16bf bf[4];
#pragma unroll
    for (int j = 0; j < 4; ++j)
      bf[j] = pa_frag_b(&sB[0][0] + (wn + j * 16) * 32, lane);

#pragma unroll
    for (int i = 0; i < 2; ++i)
#pragma unroll
      for (int j = 0; j < 4; ++j)
        acc[i][j] = __builtin_amdgcn_wmma_f32_16x16x32_bf16(
            false, af[i], false, bf[j], (short)0, acc[i][j], false, false);
    __syncthreads();
  }
#endif

  // Epilogue: C layout -> VGPR r : row = r + (lane>>4)*8, col = lane&15
#pragma unroll
  for (int i = 0; i < 2; ++i) {
#pragma unroll
    for (int j = 0; j < 4; ++j) {
      const int col  = n0 + wn + j * 16 + (lane & 15);
      const float bv = bias[col];
      const int row0 = m0 + wm + i * 16 + (lane >> 4) * 8;
#pragma unroll
      for (int r = 0; r < 8; ++r)
        C[(size_t)(row0 + r) * ND + col] = acc[i][j][r] + bv;
    }
  }
}

// --------------------- sampled scores -> sparsity measure M ----------------
// One wave per (b,l): M[b,l] = max_s(Q[l].K[idx[l,s]]) - sum_s(...)/L
__global__ __launch_bounds__(256) void pa_sample_m(const float* __restrict__ Qf,
                                                   const float* __restrict__ Kf,
                                                   const int* __restrict__ idxs,
                                                   float* __restrict__ Mval) {
  const int lane = threadIdx.x & 31;
  const int wave = threadIdx.x >> 5;
  const int gid  = blockIdx.x * 8 + wave;     // row id in [0, B*L)
  const int b    = gid / NL;
  const int l    = gid % NL;

  float q[16];
  const size_t qb = ((size_t)b * NL + l) * ND;
#pragma unroll
  for (int i = 0; i < 16; ++i) q[i] = Qf[qb + lane + 32 * i];

  float mx = -3.0e38f, sm = 0.f;
  for (int s = 0; s < SK; ++s) {
    const int id = idxs[(size_t)l * SK + s];
    const size_t kb = ((size_t)b * NL + id) * ND;
    float acc = 0.f;
#pragma unroll
    for (int i = 0; i < 16; ++i) acc += q[i] * Kf[kb + lane + 32 * i];
#pragma unroll
    for (int off = 16; off; off >>= 1) acc += __shfl_xor(acc, off, 32);
    mx = fmaxf(mx, acc);
    sm += acc;
  }
  if (lane == 0) Mval[(size_t)b * NL + l] = mx - sm / (float)NL;
}

// ------------------------------ top-45 per batch ---------------------------
__global__ __launch_bounds__(256) void pa_topk(const float* __restrict__ Mval,
                                               int* __restrict__ Mtop) {
  __shared__ float vals[NL];
  __shared__ float rmax[256];
  __shared__ int   ridx[256];
  const int b   = blockIdx.x;
  const int tid = threadIdx.x;

  for (int i = tid; i < NL; i += 256) vals[i] = Mval[(size_t)b * NL + i];
  __syncthreads();

  for (int t = 0; t < SK; ++t) {
    float bv = -3.0e38f; int bi = 0;
    for (int i = tid; i < NL; i += 256) {
      float v = vals[i];
      if (v > bv || (v == bv && i < bi)) { bv = v; bi = i; }
    }
    rmax[tid] = bv; ridx[tid] = bi;
    __syncthreads();
    for (int off = 128; off; off >>= 1) {
      if (tid < off) {
        float ov = rmax[tid + off]; int oi = ridx[tid + off];
        if (ov > rmax[tid] || (ov == rmax[tid] && oi < ridx[tid])) {
          rmax[tid] = ov; ridx[tid] = oi;
        }
      }
      __syncthreads();
    }
    if (tid == 0) { Mtop[b * SK + t] = ridx[0]; vals[ridx[0]] = -3.0e38f; }
    __syncthreads();
  }
}

// ----------- masked scores + softmax + attn@V for selected rows ------------
// One block per (b,u).
__global__ __launch_bounds__(256) void pa_attn_row(const float* __restrict__ Qf,
                                                   const float* __restrict__ Kf,
                                                   const float* __restrict__ Vf,
                                                   const int* __restrict__ Mtop,
                                                   float* __restrict__ scores,
                                                   float* __restrict__ upd) {
  __shared__ float qv[ND];
  __shared__ float red[256];
  const int tid = threadIdx.x;
  const int b   = blockIdx.x / SK;
  const int u   = blockIdx.x % SK;
  const int qi  = Mtop[b * SK + u];

  for (int d = tid; d < ND; d += 256)
    qv[d] = Qf[((size_t)b * NL + qi) * ND + d];
  __syncthreads();

  float* sc = scores + ((size_t)b * SK + u) * NL;

  // Phase 1: scores (causal-masked), block max
  float lmax = -3.0e38f;
  for (int k = tid; k < NL; k += 256) {
    float acc = 0.f;
    const float* kr = Kf + ((size_t)b * NL + k) * ND;
    __builtin_prefetch(kr + 256 * ND, 0, 1);   // global_prefetch next K row
    for (int d = 0; d < ND; ++d) acc += qv[d] * kr[d];
    acc *= SCALE;
    if (k > qi) acc = -3.0e38f;   // causal: key strictly after query -> -inf
    sc[k] = acc;
    lmax = fmaxf(lmax, acc);
  }
  red[tid] = lmax;
  __syncthreads();
  for (int off = 128; off; off >>= 1) {
    if (tid < off) red[tid] = fmaxf(red[tid], red[tid + off]);
    __syncthreads();
  }
  const float gmax = red[0];
  __syncthreads();

  // Phase 2: exponentiate, block sum
  float lsum = 0.f;
  for (int k = tid; k < NL; k += 256) {
    float p = (sc[k] <= -1.0e38f) ? 0.f : __expf(sc[k] - gmax);
    sc[k] = p;
    lsum += p;
  }
  red[tid] = lsum;
  __syncthreads();
  for (int off = 128; off; off >>= 1) {
    if (tid < off) red[tid] += red[tid + off];
    __syncthreads();
  }
  const float zinv = 1.f / red[0];

  // Phase 3: upd[d] = sum_k p_k * V[k][d] / Z
  for (int d = tid; d < ND; d += 256) {
    float acc = 0.f;
    for (int k = 0; k < NL; ++k)
      acc += sc[k] * Vf[(((size_t)b * NL + k) * ND) + d];
    upd[((size_t)b * SK + u) * ND + d] = acc * zinv;
  }
}

// ------------------------- cumsum(V) -> bf16 ctx ---------------------------
__global__ __launch_bounds__(256) void pa_cumsum(const float* __restrict__ Vf,
                                                 __bf16* __restrict__ ctxb) {
  const int g = blockIdx.x * 256 + threadIdx.x;  // [0, B*D)
  const int b = g / ND;
  const int d = g % ND;
  float acc = 0.f;
  for (int l = 0; l < NL; ++l) {
    acc += Vf[((size_t)b * NL + l) * ND + d];
    ctxb[((size_t)b * NL + l) * ND + d] = (__bf16)acc;
  }
}

// --------------------- scatter upd rows into ctx ---------------------------
__global__ __launch_bounds__(256) void pa_scatter(const float* __restrict__ upd,
                                                  const int* __restrict__ Mtop,
                                                  __bf16* __restrict__ ctxb) {
  const int g   = blockIdx.x * 256 + threadIdx.x;  // [0, B*SK*D)
  const int b   = g / (SK * ND);
  const int rem = g % (SK * ND);
  const int u   = rem / ND;
  const int d   = rem % ND;
  const int row = Mtop[b * SK + u];
  ctxb[((size_t)b * NL + row) * ND + d] = (__bf16)upd[((size_t)b * SK + u) * ND + d];
}

// ---------------------------------------------------------------------------
extern "C" void kernel_launch(void* const* d_in, const int* in_sizes, int n_in,
                              void* d_out, int out_size, void* d_ws, size_t ws_size,
                              hipStream_t stream) {
  (void)in_sizes; (void)n_in; (void)out_size; (void)ws_size;

  const float* queries = (const float*)d_in[0];
  const float* Wq = (const float*)d_in[1];
  const float* bq = (const float*)d_in[2];
  const float* Wk = (const float*)d_in[3];
  const float* bk = (const float*)d_in[4];
  const float* Wv = (const float*)d_in[5];
  const float* bv = (const float*)d_in[6];
  const float* Wo = (const float*)d_in[7];
  const float* bo = (const float*)d_in[8];
  const int*   idxs = (const int*)d_in[9];
  float* out = (float*)d_out;

  // Workspace carve-up (256B aligned regions)
  char* base = (char*)d_ws;
  size_t off = 0;
  auto alloc = [&](size_t bytes) -> char* {
    char* r = base + off;
    off = (off + bytes + 255) & ~(size_t)255;
    return r;
  };
  __bf16* qb   = (__bf16*)alloc((size_t)MROWS * ND * 2);
  __bf16* wqb  = (__bf16*)alloc((size_t)ND * ND * 2);
  __bf16* wkb  = (__bf16*)alloc((size_t)ND * ND * 2);
  __bf16* wvb  = (__bf16*)alloc((size_t)ND * ND * 2);
  __bf16* wob  = (__bf16*)alloc((size_t)ND * ND * 2);
  float*  Qf   = (float*)alloc((size_t)MROWS * ND * 4);
  float*  Kf   = (float*)alloc((size_t)MROWS * ND * 4);
  float*  Vf   = (float*)alloc((size_t)MROWS * ND * 4);
  float*  Mval = (float*)alloc((size_t)NB * NL * 4);
  int*    Mtop = (int*)alloc((size_t)NB * SK * 4);
  float*  upd  = (float*)alloc((size_t)NB * SK * ND * 4);
  float*  scr  = (float*)alloc((size_t)NB * SK * NL * 4);
  __bf16* ctxb = (__bf16*)alloc((size_t)MROWS * ND * 2);

  // 1) bf16 conversions (weights keep native [out,in] layout == WMMA B layout)
  const size_t nact = (size_t)MROWS * ND;
  pa_cvt_bf16<<<(unsigned)(nact / 256), 256, 0, stream>>>(queries, qb, nact);
  const size_t nw = (size_t)ND * ND;
  pa_cvt_bf16<<<(unsigned)(nw / 256), 256, 0, stream>>>(Wq, wqb, nw);
  pa_cvt_bf16<<<(unsigned)(nw / 256), 256, 0, stream>>>(Wk, wkb, nw);
  pa_cvt_bf16<<<(unsigned)(nw / 256), 256, 0, stream>>>(Wv, wvb, nw);
  pa_cvt_bf16<<<(unsigned)(nw / 256), 256, 0, stream>>>(Wo, wob, nw);

  // 2) Q/K/V projections via WMMA (+TDM tile staging)
  dim3 gg(ND / 128, MROWS / 128);
  pa_gemm_bf16<<<gg, 256, 0, stream>>>(qb, wqb, bq, Qf);
  pa_gemm_bf16<<<gg, 256, 0, stream>>>(qb, wkb, bk, Kf);
  pa_gemm_bf16<<<gg, 256, 0, stream>>>(qb, wvb, bv, Vf);

  // 3) sampled sparsity measure M
  pa_sample_m<<<MROWS / 8, 256, 0, stream>>>(Qf, Kf, idxs, Mval);

  // 4) top-45 per batch
  pa_topk<<<NB, 256, 0, stream>>>(Mval, Mtop);

  // 5) masked softmax + attn@V for selected query rows
  pa_attn_row<<<NB * SK, 256, 0, stream>>>(Qf, Kf, Vf, Mtop, scr, upd);

  // 6) ctx = cumsum(V) -> bf16
  pa_cumsum<<<(NB * ND) / 256, 256, 0, stream>>>(Vf, ctxb);

  // 7) scatter attention rows into ctx
  pa_scatter<<<(NB * SK * ND) / 256, 256, 0, stream>>>(upd, Mtop, ctxb);

  // 8) output projection via WMMA
  pa_gemm_bf16<<<gg, 256, 0, stream>>>(ctxb, wob, bo, out);
}
